// TriangleAttention_19378892439930
// MI455X (gfx1250) — matrix operand order repaired and checked by reading, not compiled, not run on hardware
//
#include <hip/hip_runtime.h>

#define C_CH   128
#define D_HID  32
#define NH     4
#define I_DIM  384
#define J_DIM  384

typedef __attribute__((ext_vector_type(16))) _Float16 v16h;
typedef __attribute__((ext_vector_type(8)))  float        v8f;
typedef __attribute__((ext_vector_type(4)))  unsigned int u32x4;
typedef __attribute__((ext_vector_type(8)))  int          i32x8;
typedef __attribute__((ext_vector_type(4)))  int          i32x4;

#define WMMA_F16(A,B,Cc) __builtin_amdgcn_wmma_f32_16x16x32_f16(false,(A),false,(B),(short)0,(Cc),false,false)

// ---- fragment loaders (wave32; contiguous k = 16*hs + s per element s) -------------
// A: 16x32 f16 row-major source. lane r=lane&15 -> row M=r; half hs=lane>>4.
__device__ __forceinline__ v16h frag_a_f16(const _Float16* src, int ld) {
  const int lane = threadIdx.x & 31, r = lane & 15, hs = lane >> 4;
  v16h a;
#pragma unroll
  for (int i = 0; i < 8; ++i) {
    const int k0 = 2 * i + 16 * hs;
    a[2 * i]     = src[r * ld + k0];
    a[2 * i + 1] = src[r * ld + k0 + 1];
  }
  return a;
}
// B: 32x16 (KxN) row-major source. col c=lane&15; VGPR i holds K = 2i + 16*hs.
__device__ __forceinline__ v16h frag_b_f16(const _Float16* src, int ld) {
  const int lane = threadIdx.x & 31, c = lane & 15, hs = lane >> 4;
  v16h b;
#pragma unroll
  for (int i = 0; i < 8; ++i) {
    const int k0 = 2 * i + 16 * hs;
    b[2 * i]     = src[k0 * ld + c];
    b[2 * i + 1] = src[(k0 + 1) * ld + c];
  }
  return b;
}
// B from a row-major [16 x 32] (N x K) source used transposed (e.g. K^T).
__device__ __forceinline__ v16h frag_bt_f16(const _Float16* src, int ld) {
  const int lane = threadIdx.x & 31, c = lane & 15, hs = lane >> 4;
  v16h b;
#pragma unroll
  for (int i = 0; i < 8; ++i) {
    const int k0 = 2 * i + 16 * hs;
    b[2 * i]     = src[c * ld + k0];
    b[2 * i + 1] = src[c * ld + k0 + 1];
  }
  return b;
}

// ============================================================================
// Kernel 0: pack the five f32 weight matrices [128x128] into f16 B-fragments,
// fragment-major: [frag = (w*4+kt)*8+nt][lane][16 f16]. One frag per block.
// Each GEMM B-fragment load then becomes one aligned 32-byte vector load.
// ============================================================================
__global__ __launch_bounds__(32) void k_pack(
    const float* __restrict__ wq, const float* __restrict__ wk,
    const float* __restrict__ wv, const float* __restrict__ wg,
    const float* __restrict__ wo, _Float16* __restrict__ out)
{
  const float* Ws[5] = { wq, wk, wv, wg, wo };
  const int f  = blockIdx.x;           // 0..159
  const int w  = f >> 5, rem = f & 31;
  const int kt = rem >> 3, nt = rem & 7;
  const int lane = threadIdx.x & 31, c = lane & 15, hs = lane >> 4;
  const float* src = Ws[w] + (long)kt * 32 * 128 + nt * 16 + c;
  _Float16* dst = out + ((long)f * 32 + lane) * 16;
#pragma unroll
  for (int i = 0; i < 8; ++i) {
    const int k0 = 2 * i + 16 * hs;
    dst[2 * i]     = (_Float16)src[(long)k0 * 128];
    dst[2 * i + 1] = (_Float16)src[(long)(k0 + 1) * 128];
  }
}

__device__ __forceinline__ v16h frag_b_packed(const _Float16* wpack, int frag) {
  const int lane = threadIdx.x & 31;
  return *(const v16h*)(wpack + ((long)frag * 32 + lane) * 16);
}

// ============================================================================
// Kernel 1: LayerNorm + q/k/v/gate projections (WMMA) + triangle bias.
// ============================================================================
__global__ __launch_bounds__(256) void k_ln_proj(
    const float* __restrict__ x, const float* __restrict__ ln_g,
    const float* __restrict__ ln_b, const float* __restrict__ w_tri,
    const _Float16* __restrict__ wpack, const float* __restrict__ bg,
    _Float16* __restrict__ qo, _Float16* __restrict__ ko,
    _Float16* __restrict__ vo, _Float16* __restrict__ go,
    float* __restrict__ trio)
{
  __shared__ float    xf[16][128];
  __shared__ float    red1[16][16], red2[16][16];
  __shared__ float    mu_s[16], rs_s[16];
  __shared__ _Float16 xn[16][136];

  const int tid  = threadIdx.x;
  const int pix0 = blockIdx.x * 16;
  const int i_row = pix0 / J_DIM;
  const int j0    = pix0 % J_DIM;

  { // partial LN stats: 16 threads per pixel
    const int p = tid >> 4, l = tid & 15;
    float s1 = 0.f, s2 = 0.f;
    for (int c = l; c < C_CH; c += 16) {
      float v = x[(long)(pix0 + p) * C_CH + c];
      xf[p][c] = v; s1 += v; s2 += v * v;
    }
    red1[p][l] = s1; red2[p][l] = s2;
  }
  __syncthreads();
  if (tid < 16) {
    float s1 = 0.f, s2 = 0.f;
    for (int l = 0; l < 16; ++l) { s1 += red1[tid][l]; s2 += red2[tid][l]; }
    const float mu  = s1 * (1.f / 128.f);
    const float var = s2 * (1.f / 128.f) - mu * mu;
    mu_s[tid] = mu; rs_s[tid] = rsqrtf(var + 1e-5f);
  }
  __syncthreads();
  { const int p = tid >> 4, l = tid & 15;
    const float mu = mu_s[p], rs = rs_s[p];
    for (int c = l; c < C_CH; c += 16)
      xn[p][c] = (_Float16)((xf[p][c] - mu) * rs * ln_g[c] + ln_b[c]);
  }
  __syncthreads();

  if (tid < 64) { // triangle bias: 16 pixels x 4 heads
    const int p = tid >> 2, hh = tid & 3;
    float acc = 0.f;
    for (int c = 0; c < C_CH; ++c) acc += (float)xn[p][c] * w_tri[c * NH + hh];
    trio[((long)hh * I_DIM + i_row) * J_DIM + (j0 + p)] = acc;
  }

  const int wave = tid >> 5, lane = tid & 31, r = lane & 15, hs = lane >> 4;
#pragma unroll
  for (int sel = 0; sel < 4; ++sel) {
    v8f acc = {};
#pragma unroll
    for (int ks = 0; ks < 4; ++ks) {
      v16h a = frag_a_f16(&xn[0][0] + ks * 32, 136);
      v16h b = frag_b_packed(wpack, (sel * 4 + ks) * 8 + wave);
      acc = WMMA_F16(a, b, acc);
    }
    const int n = wave * 16 + r, h = n >> 5, d = n & 31;
#pragma unroll
    for (int jv = 0; jv < 8; ++jv) {
      const int m = jv + 8 * hs;
      const long off = ((long)(i_row * NH + h) * J_DIM + (j0 + m)) * D_HID + d;
      const float v = acc[jv];
      if (sel == 0)      qo[off] = (_Float16)(v * 0.17677669529663687f); // 1/sqrt(32)
      else if (sel == 1) ko[off] = (_Float16)v;
      else if (sel == 2) vo[off] = (_Float16)v;
      else               go[off] = (_Float16)(1.f / (1.f + __expf(-(v + bg[n]))));
    }
  }
}

// ============================================================================
// Kernel 2: flash attention per (i, h). K/V staged into LDS (ld = 36 halves)
// by the Tensor Data Mover (2D tile 32x384 f16 with 2-DWORD pad every
// 16 DWORDs == ld 36) issued by wave 0, fenced by s_wait_tensorcnt.
// A cooperative copy backs the same LDS bytes so the stores are visible to
// the optimizer (no undef propagation into the WMMA chain).
// ============================================================================
__global__ __launch_bounds__(256) void k_attn(
    const _Float16* __restrict__ qi, const _Float16* __restrict__ ki,
    const _Float16* __restrict__ vi, const _Float16* __restrict__ gi,
    const float* __restrict__ trib, const float* __restrict__ mask,
    _Float16* __restrict__ og)
{
  __shared__ _Float16 Kl[J_DIM * 36];   // LDS offset 0      (alloc order)
  __shared__ _Float16 Vl[J_DIM * 36];   // LDS offset 27648
  __shared__ _Float16 Pl[8][16 * 36];

  const int tid   = threadIdx.x;
  const int i_row = blockIdx.x >> 2, h = blockIdx.x & 3;
  const long base = (long)(i_row * NH + h) * J_DIM * D_HID;
  const int wave = tid >> 5, lane = tid & 31, r = lane & 15, hs = lane >> 4;

  if (wave == 0) {
    // D# group1: data_size=2B(1), pad_enable, pad_interval=16 DW(3),
    // pad_amount=2 DW(1); tensor_dim0=32, tensor_dim1=384, tile=32x384,
    // tensor_dim0_stride=32.
    const i32x8 g1 = { (int)((1u << 16) | (1u << 20) | (3u << 22) | (1u << 25)),
                       (int)(32u << 16), (int)(384u << 16), (int)(32u << 16),
                       384, 32, 0, 0 };
    const i32x4 z4 = { 0, 0, 0, 0 };
    const i32x8 z8 = { 0, 0, 0, 0, 0, 0, 0, 0 };
    unsigned long long gaK = (unsigned long long)(const void*)(ki + base);
    unsigned long long gaV = (unsigned long long)(const void*)(vi + base);
    const u32x4 g0K = { 1u, 0u, (unsigned)gaK,
                        ((unsigned)(gaK >> 32) & 0x01FFFFFFu) | 0x80000000u };
    const u32x4 g0V = { 1u, 27648u, (unsigned)gaV,
                        ((unsigned)(gaV >> 32) & 0x01FFFFFFu) | 0x80000000u };
    __builtin_amdgcn_tensor_load_to_lds(g0K, g1, z4, z4, z8, 0);
    __builtin_amdgcn_tensor_load_to_lds(g0V, g1, z4, z4, z8, 0);
    __builtin_amdgcn_s_wait_tensorcnt(0);
  }
  // cooperative backing copy (same bytes; keeps LDS stores visible to opt)
  for (int idx = tid; idx < J_DIM * D_HID; idx += 256) {
    const int j = idx >> 5, d = idx & 31;
    Kl[j * 36 + d] = ki[base + idx];
    Vl[j * 36 + d] = vi[base + idx];
  }
  __syncthreads();

  const float* triBase = trib + (long)h * I_DIM * J_DIM;
  const float* maskRow = mask + (long)i_row * J_DIM;

  for (int t = 0; t < 3; ++t) {
    const int q0 = (wave + 8 * t) * 16;
    v16h aQ = frag_a_f16(qi + base + (long)q0 * D_HID, D_HID);
    float mrow[8], lrow[8];
    v8f o0 = {}, o1 = {};
#pragma unroll
    for (int jv = 0; jv < 8; ++jv) { mrow[jv] = -1e30f; lrow[jv] = 0.f; }

    for (int kc = 0; kc < 12; ++kc) {
      const int kg = kc * 32;
      v16h b0 = frag_bt_f16(Kl + kg * 36, 36);
      v16h b1 = frag_bt_f16(Kl + (kg + 16) * 36, 36);
      v8f  zc = {};
      v8f  s0 = WMMA_F16(aQ, b0, zc);
      v8f  s1 = WMMA_F16(aQ, b1, zc);
      const float mb0 = -1e9f * (1.f - maskRow[kg + r]);
      const float mb1 = -1e9f * (1.f - maskRow[kg + 16 + r]);
#pragma unroll
      for (int jv = 0; jv < 8; ++jv) {
        const int q = q0 + jv + 8 * hs;
        const float* triR = triBase + (long)q * J_DIM + kg;
        s0[jv] += mb0 + triR[r];
        s1[jv] += mb1 + triR[16 + r];
        float rm = fmaxf(s0[jv], s1[jv]);
        rm = fmaxf(rm, __shfl_xor(rm, 1, 32));
        rm = fmaxf(rm, __shfl_xor(rm, 2, 32));
        rm = fmaxf(rm, __shfl_xor(rm, 4, 32));
        rm = fmaxf(rm, __shfl_xor(rm, 8, 32));
        const float nm    = fmaxf(mrow[jv], rm);
        const float alpha = __expf(mrow[jv] - nm);
        mrow[jv] = nm;
        const float p0 = __expf(s0[jv] - nm);
        const float p1 = __expf(s1[jv] - nm);
        float ps = p0 + p1;
        ps += __shfl_xor(ps, 1, 32);
        ps += __shfl_xor(ps, 2, 32);
        ps += __shfl_xor(ps, 4, 32);
        ps += __shfl_xor(ps, 8, 32);
        lrow[jv] = lrow[jv] * alpha + ps;
        o0[jv] *= alpha; o1[jv] *= alpha;
        const int m = jv + 8 * hs;
        Pl[wave][m * 36 + r]      = (_Float16)p0;  // C-frag -> A-frag relayout
        Pl[wave][m * 36 + 16 + r] = (_Float16)p1;  // via wave-private LDS
      }
      v16h aP  = frag_a_f16(&Pl[wave][0], 36);
      v16h bv0 = frag_b_f16(Vl + kg * 36, 36);
      v16h bv1 = frag_b_f16(Vl + kg * 36 + 16, 36);
      o0 = WMMA_F16(aP, bv0, o0);
      o1 = WMMA_F16(aP, bv1, o1);
    }
#pragma unroll
    for (int jv = 0; jv < 8; ++jv) {
      const int q = q0 + jv + 8 * hs;
      const float inv = 1.f / lrow[jv];
      const float g0 = (float)gi[base + (long)q * D_HID + r];
      const float g1 = (float)gi[base + (long)q * D_HID + 16 + r];
      const long po = (long)(i_row * J_DIM + q) * (NH * D_HID) + h * D_HID;
      og[po + r]      = (_Float16)(o0[jv] * inv * g0);
      og[po + 16 + r] = (_Float16)(o1[jv] * inv * g1);
    }
  }
}

// ============================================================================
// Kernel 3: output projection [16 x 128] x [128 x 128] + bo -> f32 out.
// ============================================================================
__global__ __launch_bounds__(256) void k_outproj(
    const _Float16* __restrict__ og, const _Float16* __restrict__ wpack,
    const float* __restrict__ bo, float* __restrict__ out)
{
  __shared__ _Float16 al[16 * 136];
  const int tid  = threadIdx.x;
  const int pix0 = blockIdx.x * 16;
  for (int idx = tid; idx < 2048; idx += 256) {
    const int p = idx >> 7, c = idx & 127;
    al[p * 136 + c] = og[(long)(pix0 + p) * C_CH + c];
  }
  __syncthreads();
  const int wave = tid >> 5, lane = tid & 31, r = lane & 15, hs = lane >> 4;
  v8f acc = {};
#pragma unroll
  for (int ks = 0; ks < 4; ++ks) {
    v16h a = frag_a_f16(al + ks * 32, 136);
    v16h b = frag_b_packed(wpack, 128 + ks * 8 + wave);  // wo frags start at 128
    acc = WMMA_F16(a, b, acc);
  }
  const int n = wave * 16 + r;
  const float bias = bo[n];
#pragma unroll
  for (int jv = 0; jv < 8; ++jv) {
    const int m = jv + 8 * hs;
    out[(long)(pix0 + m) * C_CH + n] = acc[jv] + bias;
  }
}

// ============================================================================
extern "C" void kernel_launch(void* const* d_in, const int* in_sizes, int n_in,
                              void* d_out, int out_size, void* d_ws, size_t ws_size,
                              hipStream_t stream) {
  (void)in_sizes; (void)n_in; (void)out_size; (void)ws_size;
  const float* x     = (const float*)d_in[0];
  const float* mask  = (const float*)d_in[1];
  const float* ln_g  = (const float*)d_in[2];
  const float* ln_b  = (const float*)d_in[3];
  const float* w_tri = (const float*)d_in[4];
  const float* wq    = (const float*)d_in[5];
  const float* wk    = (const float*)d_in[6];
  const float* wv    = (const float*)d_in[7];
  const float* wg    = (const float*)d_in[8];
  const float* bg    = (const float*)d_in[9];
  const float* wo    = (const float*)d_in[10];
  const float* bo    = (const float*)d_in[11];
  float* out = (float*)d_out;

  char* ws = (char*)d_ws;
  _Float16* wpack = (_Float16*)ws;            // 160 frags * 512 f16 = 160 KiB
  ws += (size_t)160 * 32 * 16 * sizeof(_Float16);
  const size_t SZ_T = (size_t)I_DIM * NH * J_DIM * D_HID * sizeof(_Float16);
  _Float16* qw = (_Float16*)ws; ws += SZ_T;
  _Float16* kw = (_Float16*)ws; ws += SZ_T;
  _Float16* vw = (_Float16*)ws; ws += SZ_T;
  _Float16* gw = (_Float16*)ws; ws += SZ_T;
  float*  triw = (float*)ws;    ws += (size_t)NH * I_DIM * J_DIM * sizeof(float);
  _Float16* ogw = (_Float16*)ws;

  const int NPIX = I_DIM * J_DIM;
  k_pack<<<160, 32, 0, stream>>>(wq, wk, wv, wg, wo, wpack);
  k_ln_proj<<<NPIX / 16, 256, 0, stream>>>(x, ln_g, ln_b, w_tri, wpack, bg,
                                           qw, kw, vw, gw, triw);
  k_attn<<<I_DIM * NH, 256, 0, stream>>>(qw, kw, vw, gw, triw, mask, ogw);
  k_outproj<<<NPIX / 16, 256, 0, stream>>>(ogw, wpack, bo, out);
}
